// ConservativeMPLayer_23974507446373
// MI455X (gfx1250) — matrix operand order repaired and checked by reading, not compiled, not run on hardware
//
#include <hip/hip_runtime.h>
#include <math.h>

typedef _Float16 v16h __attribute__((ext_vector_type(16)));
typedef _Float16 h2v  __attribute__((ext_vector_type(2)));
typedef float    v8f  __attribute__((ext_vector_type(8)));

#define NW  4      // waves per block
#define TPB 128

// Branch-free fast GELU: x * sigmoid(1.702 x) via native exp2/rcp (no EXEC divergence)
__device__ __forceinline__ float geluf(float x) {
    float t = __expf(x * -1.702f);                 // v_exp_f32 based, branch-free
    return x * __builtin_amdgcn_rcpf(1.0f + t);    // v_rcp_f32
}

// ---------------------------------------------------------------------------
// Stage a K x N f32 weight matrix into LDS as pre-swizzled f16 WMMA B-fragments.
// Fragment f covers K-chunk kc (32 rows) and N-chunk nc (16 cols), f = nc*kchunks+kc.
// Lane L supplies 16 halves (32B contiguous): k = kc*32 + 16*(L>=16) + e, n = nc*16+(L&15)
// ---------------------------------------------------------------------------
__device__ void stage_B16(_Float16* dst, const float* W, int K, int N, int kchunks,
                          int tid, int nth) {
    int total = K * N;
    for (int t = tid; t < total; t += nth) {
        int f  = t >> 9;
        int u  = t & 511;
        int ln = u >> 4;
        int e  = u & 15;
        int nc = f / kchunks;
        int kc = f - nc * kchunks;
        int k  = kc * 32 + ((ln >> 4) << 4) + e;
        int n  = nc * 16 + (ln & 15);
        dst[f * 512 + ln * 16 + e] = (_Float16)W[k * N + n];
    }
}

// psi combined first layer: four 64x64 matrices concatenated along N (256 cols)
__device__ void stage_Bpsi(_Float16* dst, const float* W0, const float* W1,
                           const float* W2, const float* W3, int tid, int nth) {
    int total = 64 * 256;
    for (int t = tid; t < total; t += nth) {
        int f  = t >> 9;
        int u  = t & 511;
        int ln = u >> 4;
        int e  = u & 15;
        int nc = f >> 1;
        int kc = f & 1;
        int k  = kc * 32 + ((ln >> 4) << 4) + e;
        int n  = nc * 16 + (ln & 15);
        int blk = n >> 6, nn = n & 63;
        const float* W = (blk == 0) ? W0 : (blk == 1) ? W1 : (blk == 2) ? W2 : W3;
        dst[f * 512 + ln * 16 + e] = (_Float16)W[k * 64 + nn];
    }
}

// Swizzled offset (in halves) of activation element (row 0..15, feature k) inside a
// per-tile A-fragment buffer: chunk kc holds 512 halves; within a chunk lane L reads
// halves [L*16 .. L*16+15] as one contiguous 32B v16h (2 x ds_load_b128).
__device__ __forceinline__ int a_sw_off(int row, int k) {
    int kc = k >> 5, kk = k & 31;
    int lanep = row + (kk & 16);                // +16 for the high-lane half (k 16..31)
    int e     = kk & 15;
    return kc * 512 + lanep * 16 + e;
}

// ---------------------------------------------------------------------------
// Kernel 0: out = node_u (copy) and out[last] = dt = 0.015 * sigmoid(s)
// ---------------------------------------------------------------------------
__global__ void init_out_kernel(const float* __restrict__ u, const float* __restrict__ s,
                                float* __restrict__ out, int n5) {
    int i = blockIdx.x * blockDim.x + threadIdx.x;
    if (i < n5) out[i] = u[i];
    if (i == n5) out[i] = 0.015f * (1.0f / (1.0f + __expf(-s[0])));
}

// ---------------------------------------------------------------------------
// Kernel 1: per-node  g = phi1(h) + phi2(h),  h = phi_node(u)   (f16 table in ws)
// ---------------------------------------------------------------------------
__global__ void node_kernel(const float* __restrict__ u,
    const float* nW0, const float* nB0, const float* nW1, const float* nB1,
    const float* aW0, const float* aB0, const float* aW1, const float* aB1,
    const float* bW0, const float* bB0, const float* bW1, const float* bB1,
    _Float16* __restrict__ gout, int N) {
    int i = blockIdx.x * blockDim.x + threadIdx.x;
    if (i >= N) return;
    float x0 = u[i*5+0], x1 = u[i*5+1], x2 = u[i*5+2], x3 = u[i*5+3], x4 = u[i*5+4];
    float h1[64], h[64], t[32], g[64];
    for (int j = 0; j < 64; ++j)
        h1[j] = geluf(nB0[j] + x0*nW0[j] + x1*nW0[64+j] + x2*nW0[128+j]
                             + x3*nW0[192+j] + x4*nW0[256+j]);
    for (int j = 0; j < 64; ++j) {
        float o = nB1[j];
        for (int k = 0; k < 64; ++k) o = fmaf(h1[k], nW1[k*64+j], o);
        h[j] = o;
    }
    for (int j = 0; j < 32; ++j) {
        float o = aB0[j];
        for (int k = 0; k < 64; ++k) o = fmaf(h[k], aW0[k*32+j], o);
        t[j] = geluf(o);
    }
    for (int j = 0; j < 64; ++j) {
        float o = aB1[j];
        for (int k = 0; k < 32; ++k) o = fmaf(t[k], aW1[k*64+j], o);
        g[j] = o;
    }
    for (int j = 0; j < 32; ++j) {
        float o = bB0[j];
        for (int k = 0; k < 64; ++k) o = fmaf(h[k], bW0[k*32+j], o);
        t[j] = geluf(o);
    }
    for (int j = 0; j < 64; ++j) {
        float o = bB1[j];
        for (int k = 0; k < 32; ++k) o = fmaf(t[k], bW1[k*64+j], o);
        gout[i*64+j] = (_Float16)(g[j] + o);
    }
}

// ---------------------------------------------------------------------------
// Kernel 2: per-edge WMMA pipeline. Each wave processes 2 M-tiles (32 edges)
// per iteration so every LDS B-fragment is reused twice from registers.
// ---------------------------------------------------------------------------
struct EdgeArgs {
    const int* src; const int* dst;
    const float* attr; const float* area;
    const h2v* g2;                                   // node g table, f16
    const float* eW0;  const float* eB0;  const float* eW1;  const float* eB1;   // phi_edge
    const float* mW0;  const float* mB0;  const float* mW1;  const float* mB1;   // phi_msg
    const float* rhoW0; const float* rhoB0; const float* rhoW1; const float* rhoB1;
    const float* eeW0;  const float* eeB0;  const float* eeW1;  const float* eeB1;
    const float* pW0;   const float* pB0;   const float* pW1;   const float* pB1;
    const float* ruW0;  const float* ruB0;  const float* ruW1;  const float* ruB1;
    const float* s;
    float* out;
    int E; int npairs;
};

__global__ __launch_bounds__(TPB, 1) void edge_kernel(EdgeArgs A) {
    // --- weights staged once per block (all f16, pre-swizzled B-fragments) ---
    __shared__ __align__(32) _Float16 sBmsg1[24 * 512];   // 96x128 : 3 Kc x 8 Nc
    __shared__ __align__(32) _Float16 sBmsg2[16 * 512];   // 128x64 : 4 x 4
    __shared__ __align__(32) _Float16 sBpsi1[32 * 512];   // 64x256 : 2 x 16
    __shared__ float sbmsg1[128], sbmsg2[64], sbpsi1[256];
    __shared__ float sWpsi2[256 * 5], sPsiB2[5];
    __shared__ float sEW0[32], sEB0[32], sEW1[32 * 32], sEB1[32];
    // --- per-wave swizzled activation tiles (2 tiles each) + edge metadata ---
    __shared__ __align__(32) _Float16 sBufA[NW * 2 * 1536];  // v|eps (96) then m_ij (64)
    __shared__ __align__(32) _Float16 sBufB[NW * 2 * 2048];  // msg hidden (128)
    __shared__ int sSrc[NW * 32], sDst[NW * 32];

    const int tid = threadIdx.x;
    stage_B16(sBmsg1, A.mW0, 96, 128, 3, tid, TPB);
    stage_B16(sBmsg2, A.mW1, 128, 64, 4, tid, TPB);
    stage_Bpsi(sBpsi1, A.rhoW0, A.eeW0, A.pW0, A.ruW0, tid, TPB);
    for (int t = tid; t < 128; t += TPB) sbmsg1[t] = A.mB0[t];
    for (int t = tid; t < 64;  t += TPB) sbmsg2[t] = A.mB1[t];
    for (int t = tid; t < 256; t += TPB) {
        int blk = t >> 6, nn = t & 63;
        sbpsi1[t] = (blk==0) ? A.rhoB0[nn] : (blk==1) ? A.eeB0[nn]
                  : (blk==2) ? A.pB0[nn]   : A.ruB0[nn];
    }
    for (int t = tid; t < 256 * 5; t += TPB) {
        int n = t / 5, c = t - n * 5;
        int blk = n >> 6, nn = n & 63;
        float v = 0.0f;
        if      (blk == 0 && c == 0) v = A.rhoW1[nn];
        else if (blk == 1 && c == 1) v = A.eeW1[nn];
        else if (blk == 2 && c == 2) v = A.pW1[nn];
        else if (blk == 3 && c == 3) v = A.ruW1[nn * 2];
        else if (blk == 3 && c == 4) v = A.ruW1[nn * 2 + 1];
        sWpsi2[t] = v;
    }
    if (tid == 0) {
        sPsiB2[0] = A.rhoB1[0]; sPsiB2[1] = A.eeB1[0]; sPsiB2[2] = A.pB1[0];
        sPsiB2[3] = A.ruB1[0];  sPsiB2[4] = A.ruB1[1];
    }
    for (int t = tid; t < 32;      t += TPB) { sEW0[t] = A.eW0[t]; sEB0[t] = A.eB0[t]; sEB1[t] = A.eB1[t]; }
    for (int t = tid; t < 32 * 32; t += TPB) sEW1[t] = A.eW1[t];
    __syncthreads();

    const float dt = 0.015f * (1.0f / (1.0f + __expf(-A.s[0])));
    const int lane = tid & 31;
    const int w    = tid >> 5;
    const int hi8  = (lane & 16) ? 8 : 0;
    const int col  = lane & 15;
    _Float16* bufA[2] = { sBufA + w * 3072, sBufA + w * 3072 + 1536 };
    _Float16* bufB[2] = { sBufB + w * 4096, sBufB + w * 4096 + 2048 };
    int* mSrc = sSrc + w * 32;
    int* mDst = sDst + w * 32;

    const int gwave  = blockIdx.x * NW + w;
    const int nwaves = gridDim.x * NW;

    for (int pair = gwave; pair < A.npairs; pair += nwaves) {
        const int base_e = pair << 5;         // 32 edges
        {
            int e = base_e + lane;
            int ss = 0, dd = 0;
            if (e < A.E) { ss = A.src[e]; dd = A.dst[e]; }
            mSrc[lane] = ss; mDst[lane] = dd;
        }
        // gather v = g[src] + g[dst] (f16 pk adds) into K=[0,64) of each tile (swizzled)
        for (int t = lane; t < 1024; t += 32) {
            int e = t >> 5, c = t & 31;                  // edge 0..31, half2 index
            h2v v = A.g2[mSrc[e] * 32 + c] + A.g2[mDst[e] * 32 + c];
            *(h2v*)(bufA[e >> 4] + a_sw_off(e & 15, c << 1)) = v;
        }
        // eps = phi_edge(r) into K=[64,96): each lane does 2 edges x 16 outputs
#pragma unroll
        for (int t2 = 0; t2 < 2; ++t2) {
            int eg = base_e + t2 * 16 + col; eg = (eg < A.E) ? eg : (A.E - 1);
            float rv = A.attr[eg * 3 + 2];
            float th[32];
#pragma unroll
            for (int k = 0; k < 32; ++k) th[k] = geluf(fmaf(rv, sEW0[k], sEB0[k]));
            int jb = (lane >> 4) << 4;
#pragma unroll
            for (int j = 0; j < 16; ++j) {
                int jj = jb + j;
                float o = sEB1[jj];
#pragma unroll
                for (int k = 0; k < 32; ++k) o = fmaf(th[k], sEW1[k * 32 + jj], o);
                bufA[t2][a_sw_off(col, 64 + jj)] = (_Float16)o;
            }
        }
        // ---- phi_msg layer1: (16x96)@(96x128), gelu -> bufB --------------------
        v16h a0[2], a1[2], a2[2];
#pragma unroll
        for (int t2 = 0; t2 < 2; ++t2) {
            a0[t2] = *(const v16h*)(bufA[t2] + 0 * 512 + lane * 16);
            a1[t2] = *(const v16h*)(bufA[t2] + 1 * 512 + lane * 16);
            a2[t2] = *(const v16h*)(bufA[t2] + 2 * 512 + lane * 16);
        }
        for (int nc = 0; nc < 8; ++nc) {
            v16h b0 = *(const v16h*)(sBmsg1 + (nc * 3 + 0) * 512 + lane * 16);
            v16h b1 = *(const v16h*)(sBmsg1 + (nc * 3 + 1) * 512 + lane * 16);
            v16h b2 = *(const v16h*)(sBmsg1 + (nc * 3 + 2) * 512 + lane * 16);
            float bias = sbmsg1[nc * 16 + col];
#pragma unroll
            for (int t2 = 0; t2 < 2; ++t2) {
                v8f acc;
#pragma unroll
                for (int i = 0; i < 8; ++i) acc[i] = bias;
                acc = __builtin_amdgcn_wmma_f32_16x16x32_f16(false, a0[t2], false, b0, (short)0, acc, false, false);
                acc = __builtin_amdgcn_wmma_f32_16x16x32_f16(false, a1[t2], false, b1, (short)0, acc, false, false);
                acc = __builtin_amdgcn_wmma_f32_16x16x32_f16(false, a2[t2], false, b2, (short)0, acc, false, false);
#pragma unroll
                for (int r = 0; r < 8; ++r)
                    bufB[t2][a_sw_off(r + hi8, nc * 16 + col)] = (_Float16)geluf(acc[r]);
            }
        }
        // ---- phi_msg layer2: (16x128)@(128x64) -> m_ij in bufA (no activation) -
        v16h m0[2], m1[2], m2[2], m3[2];
#pragma unroll
        for (int t2 = 0; t2 < 2; ++t2) {
            m0[t2] = *(const v16h*)(bufB[t2] + 0 * 512 + lane * 16);
            m1[t2] = *(const v16h*)(bufB[t2] + 1 * 512 + lane * 16);
            m2[t2] = *(const v16h*)(bufB[t2] + 2 * 512 + lane * 16);
            m3[t2] = *(const v16h*)(bufB[t2] + 3 * 512 + lane * 16);
        }
        for (int nc = 0; nc < 4; ++nc) {
            v16h b0 = *(const v16h*)(sBmsg2 + (nc * 4 + 0) * 512 + lane * 16);
            v16h b1 = *(const v16h*)(sBmsg2 + (nc * 4 + 1) * 512 + lane * 16);
            v16h b2 = *(const v16h*)(sBmsg2 + (nc * 4 + 2) * 512 + lane * 16);
            v16h b3 = *(const v16h*)(sBmsg2 + (nc * 4 + 3) * 512 + lane * 16);
            float bias = sbmsg2[nc * 16 + col];
#pragma unroll
            for (int t2 = 0; t2 < 2; ++t2) {
                v8f acc;
#pragma unroll
                for (int i = 0; i < 8; ++i) acc[i] = bias;
                acc = __builtin_amdgcn_wmma_f32_16x16x32_f16(false, m0[t2], false, b0, (short)0, acc, false, false);
                acc = __builtin_amdgcn_wmma_f32_16x16x32_f16(false, m1[t2], false, b1, (short)0, acc, false, false);
                acc = __builtin_amdgcn_wmma_f32_16x16x32_f16(false, m2[t2], false, b2, (short)0, acc, false, false);
                acc = __builtin_amdgcn_wmma_f32_16x16x32_f16(false, m3[t2], false, b3, (short)0, acc, false, false);
#pragma unroll
                for (int r = 0; r < 8; ++r)
                    bufA[t2][a_sw_off(r + hi8, nc * 16 + col)] = (_Float16)acc[r];
            }
        }
        // ---- psi heads layer1 (16x64)@(64x256) + gelu + fused 2nd-layer dots ---
        v16h p0[2], p1[2];
#pragma unroll
        for (int t2 = 0; t2 < 2; ++t2) {
            p0[t2] = *(const v16h*)(bufA[t2] + 0 * 512 + lane * 16);
            p1[t2] = *(const v16h*)(bufA[t2] + 1 * 512 + lane * 16);
        }
        float acc2[2][5][8];
#pragma unroll
        for (int t2 = 0; t2 < 2; ++t2)
#pragma unroll
            for (int c = 0; c < 5; ++c)
#pragma unroll
                for (int r = 0; r < 8; ++r) acc2[t2][c][r] = 0.0f;
        for (int nc = 0; nc < 16; ++nc) {
            v16h b0 = *(const v16h*)(sBpsi1 + (nc * 2 + 0) * 512 + lane * 16);
            v16h b1 = *(const v16h*)(sBpsi1 + (nc * 2 + 1) * 512 + lane * 16);
            float bias = sbpsi1[nc * 16 + col];
            float wc[5];
#pragma unroll
            for (int c = 0; c < 5; ++c) wc[c] = sWpsi2[(nc * 16 + col) * 5 + c];
#pragma unroll
            for (int t2 = 0; t2 < 2; ++t2) {
                v8f d;
#pragma unroll
                for (int i = 0; i < 8; ++i) d[i] = bias;
                d = __builtin_amdgcn_wmma_f32_16x16x32_f16(false, p0[t2], false, b0, (short)0, d, false, false);
                d = __builtin_amdgcn_wmma_f32_16x16x32_f16(false, p1[t2], false, b1, (short)0, d, false, false);
#pragma unroll
                for (int r = 0; r < 8; ++r) {
                    float h = geluf(d[r]);
#pragma unroll
                    for (int c = 0; c < 5; ++c) acc2[t2][c][r] = fmaf(h, wc[c], acc2[t2][c][r]);
                }
            }
        }
        // ---- per-tile: butterfly-reduce columns, compute fluxes, scatter -------
#pragma unroll
        for (int t2 = 0; t2 < 2; ++t2) {
#pragma unroll
            for (int m = 1; m < 16; m <<= 1)
#pragma unroll
                for (int c = 0; c < 5; ++c)
#pragma unroll
                    for (int r = 0; r < 8; ++r)
                        acc2[t2][c][r] += __shfl_xor(acc2[t2][c][r], m, 32);
            float o0 = 0, o1 = 0, o2 = 0, o3 = 0, o4 = 0;
            int rsel = lane & 7;
#pragma unroll
            for (int r = 0; r < 8; ++r)
                if (rsel == r) { o0 = acc2[t2][0][r]; o1 = acc2[t2][1][r]; o2 = acc2[t2][2][r];
                                 o3 = acc2[t2][3][r]; o4 = acc2[t2][4][r]; }
            o0 += sPsiB2[0]; o1 += sPsiB2[1]; o2 += sPsiB2[2];
            o3 += sPsiB2[3]; o4 += sPsiB2[4];
            int el = t2 * 16 + (lane & 7) + hi8;
            int e  = base_e + el;
            if ((col < 8) && (e < A.E)) {
                int si = mSrc[el], di = mDst[el];
                if (si < di) {
                    float dx = A.attr[e * 3], dy = A.attr[e * 3 + 1], rr = A.attr[e * 3 + 2];
                    float inv = 1.0f / (rr + 1e-12f);
                    float nx = dx * inv, ny = dy * inv;
                    float nn = nx * nx + ny * ny;
                    float face = rr;
                    float frho = o0 * nn * face, fe = o1 * nn * face, fp = o2 * nn * face;
                    float Fx = o3 * nx - o4 * ny;     // F = a0*n + a1*t, t = (-ny, nx)
                    float Fy = o3 * ny + o4 * nx;
                    float fmx = Fx * face, fmy = Fy * face;
                    float ias = dt / A.area[si], iad = dt / A.area[di];
                    float* od = A.out;
                    atomicAdd(od + si * 5 + 0, -frho * ias); atomicAdd(od + di * 5 + 0, frho * iad);
                    atomicAdd(od + si * 5 + 1, -fe * ias);   atomicAdd(od + di * 5 + 1, fe * iad);
                    atomicAdd(od + si * 5 + 2, -fp * ias);   atomicAdd(od + di * 5 + 2, fp * iad);
                    atomicAdd(od + si * 5 + 3, -fmx * ias);  atomicAdd(od + di * 5 + 3, fmx * iad);
                    atomicAdd(od + si * 5 + 4, -fmy * ias);  atomicAdd(od + di * 5 + 4, fmy * iad);
                }
            }
        }
    }
}

// ---------------------------------------------------------------------------
extern "C" void kernel_launch(void* const* d_in, const int* in_sizes, int n_in,
                              void* d_out, int out_size, void* d_ws, size_t ws_size,
                              hipStream_t stream) {
    const float* node_u = (const float*)d_in[0];
    const int*   eidx   = (const int*)d_in[1];
    const float* attr   = (const float*)d_in[2];
    const float* area   = (const float*)d_in[3];
    // param flattening order: insertion order (phi_node first, W0 size 320) vs
    // jax-pytree sorted keys (phi1 first, W0 size 2048)
    bool ins = (in_sizes[4] == 320);
    int b_node = ins ? 4  : 20;
    int b_p1   = ins ? 8  : 4;
    int b_p2   = ins ? 12 : 8;
    int b_edge = ins ? 16 : 12;
    int b_msg  = ins ? 20 : 16;
    int b_rho  = ins ? 24 : 32;
    int b_e    = ins ? 28 : 24;
    int b_p    = ins ? 32 : 28;
    int b_ru   = 36;
    const float* s = (const float*)d_in[40];

    int N = in_sizes[0] / 5;
    int E = in_sizes[1] / 2;
    float* out = (float*)d_out;
    _Float16* g = (_Float16*)d_ws;

#define P(b, k) ((const float*)d_in[(b) + (k)])

    int n5 = N * 5;
    init_out_kernel<<<(n5 + 256) / 256, 256, 0, stream>>>(node_u, s, out, n5);

    node_kernel<<<(N + 255) / 256, 256, 0, stream>>>(node_u,
        P(b_node,0), P(b_node,1), P(b_node,2), P(b_node,3),
        P(b_p1,0),   P(b_p1,1),   P(b_p1,2),   P(b_p1,3),
        P(b_p2,0),   P(b_p2,1),   P(b_p2,2),   P(b_p2,3),
        g, N);

    EdgeArgs a;
    a.src = eidx; a.dst = eidx + E; a.attr = attr; a.area = area;
    a.g2 = (const h2v*)g;
    a.eW0 = P(b_edge,0); a.eB0 = P(b_edge,1); a.eW1 = P(b_edge,2); a.eB1 = P(b_edge,3);
    a.mW0 = P(b_msg,0);  a.mB0 = P(b_msg,1);  a.mW1 = P(b_msg,2);  a.mB1 = P(b_msg,3);
    a.rhoW0 = P(b_rho,0); a.rhoB0 = P(b_rho,1); a.rhoW1 = P(b_rho,2); a.rhoB1 = P(b_rho,3);
    a.eeW0  = P(b_e,0);   a.eeB0  = P(b_e,1);   a.eeW1  = P(b_e,2);   a.eeB1  = P(b_e,3);
    a.pW0   = P(b_p,0);   a.pB0   = P(b_p,1);   a.pW1   = P(b_p,2);   a.pB1   = P(b_p,3);
    a.ruW0  = P(b_ru,0);  a.ruB0  = P(b_ru,1);  a.ruW1  = P(b_ru,2);  a.ruB1  = P(b_ru,3);
    a.s = s; a.out = out; a.E = E; a.npairs = (E + 31) / 32;

    edge_kernel<<<2048, TPB, 0, stream>>>(a);
#undef P
}
